// MelodyGenerator_17300128268569
// MI455X (gfx1250) — matrix operand order repaired and checked by reading, not compiled, not run on hardware
//
#include <hip/hip_runtime.h>
#include <hip/hip_bf16.h>

// ---------------------------------------------------------------------------
// Persistent cooperative 2-layer LSTM decoder for MI455X (gfx1250, wave32).
//
//   T = length*128 outer steps x 3 inner tokens, strictly sequential.
//   Each inner step:
//     layer0: gates[2048] = [Wih0|Whh0][2048x768] @ [x_s;h0][768]
//     layer1: gates[2048] = [Wih1|Whh1][2048x1024] @ [y0;h1][1024]
//     out   : [Wp;Wv][256x512] @ y1 + [bp;bv]
//
//   32 workgroups, each owns 16 hidden units (= 64 gate rows, 4 WMMA M-tiles).
//   Weights pre-packed to bf16 WMMA A-fragment layout (L2-resident, ~7.3MB).
//   Matvec: A = 16x32 weight tile, B = input vector replicated across all 16
//   N columns -> every column of the 16x16 f32 D tile equals the 16 outputs;
//   K accumulates through the chained v_wmma accumulator.
//   h vectors: bf16, ping-pong double buffered in global ws.
//   Grid sync: monotonically increasing atomic counter, 2 barriers/step,
//   release via __threadfence before arrival, acquire via per-thread
//   agent-scope fence (L0 invalidate) after completion.
// ---------------------------------------------------------------------------

typedef __attribute__((ext_vector_type(16))) __bf16 v16bf;
typedef __attribute__((ext_vector_type(8)))  float  v8f;

#define NWG   32
#define TPB   256
#define C0    24   // K chunks layer0 (768/32)
#define C1    32   // K chunks layer1 (1024/32)
#define CP    16   // K chunks out-proj (512/32)

// workspace byte offsets
#define OFF_SYNC 0
#define OFF_X    256                       // 768 bf16  (3 x 256 tokens)
#define OFF_H0   2048                      // 2x512 bf16 (ping-pong)
#define OFF_H1   4096                      // 2x512 bf16
#define OFF_B0   8192                      // 32*64 f32 packed gate bias L0
#define OFF_B1   16384                     // 32*64 f32 packed gate bias L1
#define OFF_BO   24576                     // 256 f32 [bp;bv]
#define OFF_A0   32768                     // 32*4*24*512 bf16 = 3 MB
#define A0_HALVES (NWG*4*C0*512)
#define OFF_A1   (OFF_A0 + A0_HALVES*2)    // 32*4*32*512 bf16 = 4 MB
#define A1_HALVES (NWG*4*C1*512)
#define OFF_AP   (OFF_A1 + A1_HALVES*2)    // 16*16*512 bf16 = 256 KB
#define AP_HALVES (16*CP*512)

// K index inside a lane's 16 contiguous halves of an A-fragment
// (contiguous K order: lanes 0-15 hold K{0..15}, lanes 16-31 hold
//  K{16..31}; 2 halves per VGPR)
__device__ __host__ inline int a_kofs(int lane, int e) {
    return (lane < 16) ? e : e + 16;
}

// -------------------------- weight packing --------------------------------

__global__ void pack_gates(const float* __restrict__ Wih,
                           const float* __restrict__ Whh,
                           const float* __restrict__ bih,
                           const float* __restrict__ bhh,
                           __bf16* __restrict__ Apack,
                           float* __restrict__ bpack,
                           int Kih, int nChunks) {
    long idx = (long)blockIdx.x * TPB + threadIdx.x;
    long total = (long)NWG * 4 * nChunks * 512;
    if (idx < total) {
        int e    = (int)(idx & 15);
        int lane = (int)((idx >> 4) & 31);
        long rest = idx >> 9;
        int kc = (int)(rest % nChunks);
        int m  = (int)((rest / nChunks) & 3);        // gate tile = gate id
        int wg = (int)(rest / ((long)nChunks * 4));
        int mrow = lane & 15;
        int row  = m * 512 + wg * 16 + mrow;          // original gate row
        int k    = kc * 32 + a_kofs(lane, e);
        float v = (k < Kih) ? Wih[(long)row * Kih + k]
                            : Whh[(long)row * 512 + (k - Kih)];
        Apack[idx] = (__bf16)v;
    }
    if (idx < NWG * 64) {                             // packed combined bias
        int wg = (int)idx / 64, r = (int)idx % 64;
        int row = (r / 16) * 512 + wg * 16 + (r % 16);
        bpack[idx] = bih[row] + bhh[row];
    }
}

__global__ void pack_out(const float* __restrict__ Wp, const float* __restrict__ bp,
                         const float* __restrict__ Wv, const float* __restrict__ bv,
                         __bf16* __restrict__ APpack, float* __restrict__ bOut) {
    int idx = blockIdx.x * TPB + threadIdx.x;
    if (idx < AP_HALVES) {
        int e    = idx & 15;
        int lane = (idx >> 4) & 31;
        int rest = idx >> 9;
        int kc = rest % CP, tile = rest / CP;
        int row = tile * 16 + (lane & 15);
        int k   = kc * 32 + a_kofs(lane, e);
        float v = (row < 128) ? Wp[row * 512 + k] : Wv[(row - 128) * 512 + k];
        APpack[idx] = (__bf16)v;
    }
    if (idx < 256) bOut[idx] = (idx < 128) ? bp[idx] : bv[idx - 128];
}

__global__ void init_state(const int* __restrict__ tempo,
                           const int* __restrict__ key_sig,
                           const int* __restrict__ length,
                           const float* __restrict__ emb,
                           __bf16* __restrict__ xbuf,
                           __bf16* __restrict__ h0buf,
                           __bf16* __restrict__ h1buf,
                           int* __restrict__ sync) {
    int tid = blockIdx.x * TPB + threadIdx.x;
    if (tid == 0) *sync = 0;
    if (tid < 1024) { h0buf[tid] = (__bf16)0.f; h1buf[tid] = (__bf16)0.f; }
    if (tid < 768) {
        int s = tid / 256, col = tid % 256;
        int rowi = (s == 0) ? tempo[0] : (s == 1) ? key_sig[0] : length[0];
        xbuf[tid] = (__bf16)emb[rowi * 256 + col];
    }
}

// ----------------------------- main kernel --------------------------------

__device__ inline float sigf(float x) { return 1.f / (1.f + __expf(-x)); }

__device__ inline void grid_barrier(int* sync, int target, int tid) {
    __threadfence();                       // release: flush our h/x/out stores
    __syncthreads();
    if (tid == 0) {
        atomicAdd(sync, 1);
        while (__hip_atomic_load(sync, __ATOMIC_ACQUIRE,
                                 __HIP_MEMORY_SCOPE_AGENT) < target) {
            __builtin_amdgcn_s_sleep(1);
        }
    }
    __syncthreads();
    // acquire for EVERY wave: invalidate WGP-local caches so subsequent
    // h-vector loads observe other WGPs' stores (agent-scope INV).
    __builtin_amdgcn_fence(__ATOMIC_ACQUIRE, "agent");
}

__global__ __launch_bounds__(TPB, 1) void lstm_main(
        const __bf16* __restrict__ A0, const __bf16* __restrict__ A1,
        const __bf16* __restrict__ AP,
        const float* __restrict__ b0, const float* __restrict__ b1,
        const float* __restrict__ bOut,
        __bf16* __restrict__ xbuf, __bf16* __restrict__ h0buf,
        __bf16* __restrict__ h1buf,
        const int* __restrict__ lengthp, float* __restrict__ out,
        int* __restrict__ sync) {
    const int wg   = blockIdx.x;
    const int tid  = threadIdx.x;
    const int lane = tid & 31;
    const int wv   = tid >> 5;            // 8 wave32s
    const int m    = wv & 3;              // gate tile (i,f,g,o)
    const int kh   = wv >> 2;             // K half handled by this wave
    const int boff = (lane & 16) ? 16 : 0; // B-fragment: contiguous 16 halves

    __shared__ float part[2][64];
    __shared__ float gates[64];
    __shared__ float pout[8][16];
    __shared__ float cst[2][16];          // persistent cell state (16 units/WG)

    if (tid < 16) { cst[0][tid] = 0.f; cst[1][tid] = 0.f; }
    __syncthreads();

    const int T = lengthp[0] * 128;
    int bar = 0;
    int p = 0;                            // ping-pong parity

    const __bf16* myA0 = A0 + (size_t)wg * 4 * C0 * 512;
    const __bf16* myA1 = A1 + (size_t)wg * 4 * C1 * 512;
    const float*  myb0 = b0 + wg * 64;
    const float*  myb1 = b1 + wg * 64;

    for (int t = 0; t < T; ++t) {
        for (int s = 0; s < 3; ++s) {
            const int pn = p ^ 1;
            const __bf16* h0r = h0buf + p  * 512;
            __bf16*       h0w = h0buf + pn * 512;
            const __bf16* h1r = h1buf + p  * 512;
            __bf16*       h1w = h1buf + pn * 512;
            const __bf16* xs  = xbuf + s * 256;

            // ---- layer 0: K = 768 = [x_s(256) ; h0(512)] ----
            {
                v8f acc = {};
                #pragma unroll
                for (int j = 0; j < C0 / 2; ++j) {
                    const int kc = kh * (C0 / 2) + j;
                    const v16bf a = *(const v16bf*)(myA0 +
                        (((size_t)(m * C0 + kc)) << 9) + (lane << 4));
                    const __bf16* vb = (kc < 8) ? (xs + kc * 32)
                                                : (h0r + (kc - 8) * 32);
                    const v16bf bfr = *(const v16bf*)(vb + boff);
                    acc = __builtin_amdgcn_wmma_f32_16x16x32_bf16(
                        false, a, false, bfr, (short)0, acc, false, false);
                }
                if ((lane & 15) == 0) {
                    int hi = (lane >> 4) * 8;   // D: lanes0-15 M=r, 16-31 M=8+r
                    #pragma unroll
                    for (int r = 0; r < 8; ++r) part[kh][m * 16 + hi + r] = acc[r];
                }
            }
            __syncthreads();
            if (tid < 64) gates[tid] = part[0][tid] + part[1][tid] + myb0[tid];
            __syncthreads();
            if (tid < 16) {
                float i = gates[tid], f = gates[16 + tid];
                float g = gates[32 + tid], o = gates[48 + tid];
                float c = sigf(f) * cst[0][tid] + sigf(i) * tanhf(g);
                cst[0][tid] = c;
                h0w[wg * 16 + tid] = (__bf16)(sigf(o) * tanhf(c));
            }
            grid_barrier(sync, (++bar) * NWG, tid);   // h0/y0 ready

            // ---- layer 1: K = 1024 = [y0(512) ; h1(512)] ----
            {
                v8f acc = {};
                #pragma unroll
                for (int j = 0; j < C1 / 2; ++j) {
                    const int kc = kh * (C1 / 2) + j;
                    const v16bf a = *(const v16bf*)(myA1 +
                        (((size_t)(m * C1 + kc)) << 9) + (lane << 4));
                    const __bf16* vb = (kc < 16) ? (h0w + kc * 32)
                                                 : (h1r + (kc - 16) * 32);
                    const v16bf bfr = *(const v16bf*)(vb + boff);
                    acc = __builtin_amdgcn_wmma_f32_16x16x32_bf16(
                        false, a, false, bfr, (short)0, acc, false, false);
                }
                if ((lane & 15) == 0) {
                    int hi = (lane >> 4) * 8;
                    #pragma unroll
                    for (int r = 0; r < 8; ++r) part[kh][m * 16 + hi + r] = acc[r];
                }
            }
            __syncthreads();
            if (tid < 64) gates[tid] = part[0][tid] + part[1][tid] + myb1[tid];
            __syncthreads();
            if (tid < 16) {
                float i = gates[tid], f = gates[16 + tid];
                float g = gates[32 + tid], o = gates[48 + tid];
                float c = sigf(f) * cst[1][tid] + sigf(i) * tanhf(g);
                cst[1][tid] = c;
                h1w[wg * 16 + tid] = (__bf16)(sigf(o) * tanhf(c));
            }
            grid_barrier(sync, (++bar) * NWG, tid);   // y1 ready

            // ---- output projection: 256 x 512 (WGs 0..15, one 16-row tile) ----
            if (wg < 16) {
                v8f acc = {};
                #pragma unroll
                for (int j = 0; j < 2; ++j) {
                    const int kc = wv * 2 + j;
                    const v16bf a = *(const v16bf*)(AP +
                        (((size_t)(wg * CP + kc)) << 9) + (lane << 4));
                    const v16bf bfr = *(const v16bf*)(h1w + kc * 32 + boff);
                    acc = __builtin_amdgcn_wmma_f32_16x16x32_bf16(
                        false, a, false, bfr, (short)0, acc, false, false);
                }
                if ((lane & 15) == 0) {
                    int hi = (lane >> 4) * 8;
                    #pragma unroll
                    for (int r = 0; r < 8; ++r) pout[wv][hi + r] = acc[r];
                }
                __syncthreads();
                if (tid < 16) {
                    float v = bOut[wg * 16 + tid];
                    #pragma unroll
                    for (int w = 0; w < 8; ++w) v += pout[w][tid];
                    out[((size_t)t * 3 + s) * 256 + wg * 16 + tid] = v;
                    // out token becomes next outer step's x_s (in place: safe,
                    // x_s was last read before barrier 1 of this inner step)
                    xbuf[s * 256 + wg * 16 + tid] = (__bf16)v;
                }
            }
            p = pn;   // next inner step's barriers order these writes for readers
        }
    }
}

// ------------------------------- launcher ---------------------------------

extern "C" void kernel_launch(void* const* d_in, const int* in_sizes, int n_in,
                              void* d_out, int out_size, void* d_ws, size_t ws_size,
                              hipStream_t stream) {
    (void)in_sizes; (void)n_in; (void)out_size; (void)ws_size;
    const int*   tempo  = (const int*)d_in[0];
    const int*   keysig = (const int*)d_in[1];
    const int*   length = (const int*)d_in[2];
    const float* emb    = (const float*)d_in[3];
    const float* Wih0   = (const float*)d_in[4];
    const float* Whh0   = (const float*)d_in[5];
    const float* bih0   = (const float*)d_in[6];
    const float* bhh0   = (const float*)d_in[7];
    const float* Wih1   = (const float*)d_in[8];
    const float* Whh1   = (const float*)d_in[9];
    const float* bih1   = (const float*)d_in[10];
    const float* bhh1   = (const float*)d_in[11];
    const float* Wp     = (const float*)d_in[12];
    const float* bp     = (const float*)d_in[13];
    const float* Wv     = (const float*)d_in[14];
    const float* bv     = (const float*)d_in[15];

    char* ws = (char*)d_ws;
    int*    syncp = (int*)(ws + OFF_SYNC);
    __bf16* xbuf  = (__bf16*)(ws + OFF_X);
    __bf16* h0b   = (__bf16*)(ws + OFF_H0);
    __bf16* h1b   = (__bf16*)(ws + OFF_H1);
    float*  b0p   = (float*)(ws + OFF_B0);
    float*  b1p   = (float*)(ws + OFF_B1);
    float*  bOp   = (float*)(ws + OFF_BO);
    __bf16* A0p   = (__bf16*)(ws + OFF_A0);
    __bf16* A1p   = (__bf16*)(ws + OFF_A1);
    __bf16* APp   = (__bf16*)(ws + OFF_AP);

    pack_gates<<<(A0_HALVES + TPB - 1) / TPB, TPB, 0, stream>>>(
        Wih0, Whh0, bih0, bhh0, A0p, b0p, 256, C0);
    pack_gates<<<(A1_HALVES + TPB - 1) / TPB, TPB, 0, stream>>>(
        Wih1, Whh1, bih1, bhh1, A1p, b1p, 512, C1);
    pack_out<<<(AP_HALVES + TPB - 1) / TPB, TPB, 0, stream>>>(
        Wp, bp, Wv, bv, APp, bOp);
    init_state<<<4, TPB, 0, stream>>>(tempo, keysig, length, emb,
                                      xbuf, h0b, h1b, syncp);
    lstm_main<<<NWG, TPB, 0, stream>>>(A0p, A1p, APp, b0p, b1p, bOp,
                                       xbuf, h0b, h1b, length,
                                       (float*)d_out, syncp);
}
